// GraphDeepOnet_58514634441395
// MI455X (gfx1250) — compile-verified, matrix operand order checked by reading, not run on hardware
//
#include <hip/hip_runtime.h>
#include <hip/hip_bf16.h>
#include <hip/hip_fp16.h>

// ---------------- problem constants (from reference) ----------------
#define B_GR   32
#define NK_PER 1024
#define NQ_PER 8192
#define DEG_   16
#define NK_    (B_GR * NK_PER)     // 32768
#define NQ_    (B_GR * NQ_PER)     // 262144
#define E_     (NK_ * DEG_)        // 524288
#define H_     512
#define OUT_   3

typedef __attribute__((ext_vector_type(16))) _Float16 v16h;
typedef __attribute__((ext_vector_type(8)))  _Float16 v8h;
typedef __attribute__((ext_vector_type(8)))  float    v8f;

__device__ __forceinline__ v8f vzero8() {
  v8f z = {0.f,0.f,0.f,0.f,0.f,0.f,0.f,0.f};
  return z;
}

// Load one 16x32 f16 A-fragment row-slice (or B^T-fragment) for this lane.
// 16-bit A layout (ISA 7.12.2): lanes 0-15 hold K {0..7,16..23} of row M=lane,
// lanes 16-31 hold K {8..15,24..31}.  Two 16B chunks, combined into v16h.
__device__ __forceinline__ v16h load_frag(const _Float16* p, int hi) {
  v8h c0 = *reinterpret_cast<const v8h*>(p + hi * 8);
  v8h c1 = *reinterpret_cast<const v8h*>(p + 16 + hi * 8);
  return __builtin_shufflevector(c0, c1, 0,1,2,3,4,5,6,7,8,9,10,11,12,13,14,15);
}

__device__ __forceinline__ v8f wmma_f16(v16h a, v16h b, v8f c) {
  return __builtin_amdgcn_wmma_f32_16x16x32_f16(false, a, false, b, (short)0, c,
                                                false, false);
}

// Core K=512 loop: 2 row-tiles (A panels) x 4 col-tiles of WT (N-major f16).
// Each B fragment is loaded once and used by both A panels (1 global b128 per
// WMMA).  B fragments are explicitly double-buffered (b* live, n* prefetched
// one k-step ahead) so the scheduler keeps ~8 loads in flight under the WMMAs
// instead of collapsing them onto one register set with loadcnt==0 waits.
__device__ __forceinline__ void gemm_k512_2x4(const _Float16* __restrict__ A0,
                                              const _Float16* __restrict__ A1,
                                              const _Float16* __restrict__ WT,
                                              int cb, int lr, int hi, v8f acc[2][4]) {
  const _Float16* w = WT + (size_t)(cb + lr) * H_;
  v16h b0 = load_frag(w, hi);
  v16h b1 = load_frag(w + (size_t)16 * H_, hi);
  v16h b2 = load_frag(w + (size_t)32 * H_, hi);
  v16h b3 = load_frag(w + (size_t)48 * H_, hi);
#pragma unroll 2
  for (int k0 = 0; k0 < H_ - 32; k0 += 32) {
    // prefetch next k-step's B fragments (distinct registers -> overlap)
    v16h n0 = load_frag(w + k0 + 32, hi);
    v16h n1 = load_frag(w + (size_t)16 * H_ + k0 + 32, hi);
    v16h n2 = load_frag(w + (size_t)32 * H_ + k0 + 32, hi);
    v16h n3 = load_frag(w + (size_t)48 * H_ + k0 + 32, hi);
    v16h a0 = load_frag(A0 + k0, hi);
    v16h a1 = load_frag(A1 + k0, hi);
    acc[0][0] = wmma_f16(a0, b0, acc[0][0]);
    acc[1][0] = wmma_f16(a1, b0, acc[1][0]);
    acc[0][1] = wmma_f16(a0, b1, acc[0][1]);
    acc[1][1] = wmma_f16(a1, b1, acc[1][1]);
    acc[0][2] = wmma_f16(a0, b2, acc[0][2]);
    acc[1][2] = wmma_f16(a1, b2, acc[1][2]);
    acc[0][3] = wmma_f16(a0, b3, acc[0][3]);
    acc[1][3] = wmma_f16(a1, b3, acc[1][3]);
    b0 = n0; b1 = n1; b2 = n2; b3 = n3;
  }
  // peeled final k-step (k0 = H_-32)
  {
    v16h a0 = load_frag(A0 + H_ - 32, hi);
    v16h a1 = load_frag(A1 + H_ - 32, hi);
    acc[0][0] = wmma_f16(a0, b0, acc[0][0]);
    acc[1][0] = wmma_f16(a1, b0, acc[1][0]);
    acc[0][1] = wmma_f16(a0, b1, acc[0][1]);
    acc[1][1] = wmma_f16(a1, b1, acc[1][1]);
    acc[0][2] = wmma_f16(a0, b2, acc[0][2]);
    acc[1][2] = wmma_f16(a1, b2, acc[1][2]);
    acc[0][3] = wmma_f16(a0, b3, acc[0][3]);
    acc[1][3] = wmma_f16(a1, b3, acc[1][3]);
  }
}

// ---------------- small utility kernels ----------------
__global__ void zero_f32(float* p, long n) {
  long i = (long)blockIdx.x * blockDim.x + threadIdx.x;
  if (i < n) p[i] = 0.f;
}

// W[k*512+n] (f32, 512x512) -> WT[n*512+k] (f16)
__global__ void conv_transpose_f16(const float* __restrict__ W, _Float16* __restrict__ WT) {
  int idx = blockIdx.x * blockDim.x + threadIdx.x;   // 512*512 threads
  int n = idx >> 9, k = idx & 511;
  WT[idx] = (_Float16)W[k * H_ + n];
}

// wo2[k*3+n] (f32, 512x3) -> padded WT[n*512+k] (f16, 16x512, zeros for n>=3)
__global__ void conv_wo2_pad(const float* __restrict__ W, _Float16* __restrict__ WT) {
  int idx = blockIdx.x * blockDim.x + threadIdx.x;   // 16*512 threads
  int n = idx >> 9, k = idx & 511;
  WT[idx] = (n < OUT_) ? (_Float16)W[k * OUT_ + n] : (_Float16)0.f;
}

__global__ void deg_count(const int* __restrict__ dst, float* __restrict__ deg) {
  int e = blockIdx.x * blockDim.x + threadIdx.x;
  if (e < E_) atomicAdd(deg + dst[e], 1.0f);
}

__global__ void deg_invert(float* deg) {
  int i = blockIdx.x * blockDim.x + threadIdx.x;
  if (i < NK_) deg[i] = 1.0f / fmaxf(deg[i], 1.0f);
}

// layer-0 aggregation: m0[dst, 0:3] += known[src, 0:3]
__global__ void agg_edges_f3(const float* __restrict__ X, const int* __restrict__ src,
                             const int* __restrict__ dst, float* __restrict__ M) {
  int e = blockIdx.x * blockDim.x + threadIdx.x;
  if (e >= E_) return;
  int s = src[e], d = dst[e];
#pragma unroll
  for (int k = 0; k < 3; ++k) atomicAdd(M + d * 3 + k, X[s * 3 + k]);
}

// h0 = relu((m0/deg) @ gc_w0 + b0)  -> f16
__global__ void layer0_dense(const float* __restrict__ m0, const float* __restrict__ invdeg,
                             const float* __restrict__ W0, const float* __restrict__ b0,
                             _Float16* __restrict__ Hout) {
  int idx = blockIdx.x * blockDim.x + threadIdx.x;   // NK*512 threads
  int r = idx >> 9, c = idx & 511;
  float id = invdeg[r];
  float v = m0[r * 3 + 0] * id * W0[c]
          + m0[r * 3 + 1] * id * W0[H_ + c]
          + m0[r * 3 + 2] * id * W0[2 * H_ + c] + b0[c];
  Hout[idx] = (_Float16)fmaxf(v, 0.f);
}

// feature aggregation: M[dst, :] += H[src, :]  (one edge per block, 512 feats)
__global__ void agg_edges_f16(const _Float16* __restrict__ H, const int* __restrict__ src,
                              const int* __restrict__ dst, float* __restrict__ M) {
  int e = blockIdx.x;
  int s = src[e], d = dst[e];
  int c = threadIdx.x * 2;                           // 256 threads -> 2 cols each
  float a = (float)H[(size_t)s * H_ + c];
  float b = (float)H[(size_t)s * H_ + c + 1];
  atomicAdd(M + (size_t)d * H_ + c, a);
  atomicAdd(M + (size_t)d * H_ + c + 1, b);
}

// h = relu((M*invdeg) @ WT^T + bias)   [32 rows / block, WMMA 2x4 per wave]
__global__ void __launch_bounds__(256)
gcn_gemm_relu(const float* __restrict__ M, const float* __restrict__ invdeg,
              const _Float16* __restrict__ WT, const float* __restrict__ bias,
              _Float16* __restrict__ Hout) {
  __shared__ _Float16 As[32 * H_];
  const int r0 = blockIdx.x * 32;
  const int tid = threadIdx.x;
#pragma unroll 4
  for (int i = 0; i < 64; ++i) {
    int idx = i * 256 + tid;
    int r = idx >> 9, c = idx & 511;
    As[idx] = (_Float16)(M[(size_t)(r0 + r) * H_ + c] * invdeg[r0 + r]);
  }
  __syncthreads();
  const int wave = tid >> 5, lane = tid & 31;
  const int lr = lane & 15, hi = lane >> 4;
  const int cb = wave * 64;            // 8 waves x 64 disjoint cols
  v8f acc[2][4];
#pragma unroll
  for (int rt = 0; rt < 2; ++rt)
#pragma unroll
    for (int t = 0; t < 4; ++t) acc[rt][t] = vzero8();
  gemm_k512_2x4(As + (size_t)lr * H_, As + (size_t)(16 + lr) * H_, WT, cb, lr, hi, acc);
#pragma unroll
  for (int rt = 0; rt < 2; ++rt)
#pragma unroll
    for (int t = 0; t < 4; ++t) {
      int n = cb + t * 16 + lr;
      float bs = bias[n];
#pragma unroll
      for (int v = 0; v < 8; ++v) {
        int m = rt * 16 + hi * 8 + v;
        Hout[(size_t)(r0 + m) * H_ + n] = (_Float16)fmaxf(acc[rt][t][v] + bs, 0.f);
      }
    }
}

// pooled[b,c] = mean over 1024 nodes of h2
__global__ void pool_mean(const _Float16* __restrict__ H, float* __restrict__ pooled) {
  int idx = blockIdx.x * blockDim.x + threadIdx.x;   // B*H threads
  int b = idx >> 9, c = idx & 511;
  float s = 0.f;
  const _Float16* p = H + (size_t)b * NK_PER * H_ + c;
  for (int i = 0; i < NK_PER; ++i) s += (float)p[(size_t)i * H_];
  pooled[idx] = s * (1.0f / NK_PER);
}

// field[b,n] = pooled[b,:] @ gc_wr[:,n] + br[n]
__global__ void field_readout(const float* __restrict__ pooled, const float* __restrict__ Wr,
                              const float* __restrict__ br, float* __restrict__ field) {
  int idx = blockIdx.x * blockDim.x + threadIdx.x;   // B*H threads
  int b = idx >> 9, n = idx & 511;
  float s = br[n];
  for (int k = 0; k < H_; ++k) s += pooled[b * H_ + k] * Wr[k * H_ + n];
  field[idx] = s;
}

// Fused trunk + broadcast-multiply + out MLP over 32 query rows per block.
__global__ void __launch_bounds__(256)
trunk_fused(const float* __restrict__ nodes,
            const float* __restrict__ wt1, const float* __restrict__ bt1,
            const _Float16* __restrict__ wt2T, const float* __restrict__ bt2,
            const float* __restrict__ field,
            const _Float16* __restrict__ wo1T, const float* __restrict__ bo1,
            const _Float16* __restrict__ wo2Tp, const float* __restrict__ bo2,
            float* __restrict__ out) {
  extern __shared__ _Float16 smem[];
  _Float16* bufA = smem;               // 32*512 f16: t1, later h
  _Float16* bufZ = smem + 32 * H_;     // 32*512 f16: z
  const int r0 = blockIdx.x * 32;
  const int g = r0 >> 13;              // graph id (NQ_PER = 8192)
  const int tid = threadIdx.x;

  // stage 0: t1 = relu(nodes @ wt1 + bt1) -> bufA (f16)
#pragma unroll 4
  for (int i = 0; i < 64; ++i) {
    int idx = i * 256 + tid;
    int r = idx >> 9, c = idx & 511;
    const float* nr = nodes + (size_t)(r0 + r) * 3;
    float v = nr[0] * wt1[c] + nr[1] * wt1[H_ + c] + nr[2] * wt1[2 * H_ + c] + bt1[c];
    bufA[idx] = (_Float16)fmaxf(v, 0.f);
  }
  __syncthreads();

  const int wave = tid >> 5, lane = tid & 31;
  const int lr = lane & 15, hi = lane >> 4;
  const int cb = wave * 64;            // 8 waves x 64 disjoint cols

  // stage 1: t2 = t1 @ wt2 + bt2 ; z = field[g] * t2 -> bufZ
  {
    v8f acc[2][4];
#pragma unroll
    for (int rt = 0; rt < 2; ++rt)
#pragma unroll
      for (int t = 0; t < 4; ++t) acc[rt][t] = vzero8();
    gemm_k512_2x4(bufA + (size_t)lr * H_, bufA + (size_t)(16 + lr) * H_,
                  wt2T, cb, lr, hi, acc);
#pragma unroll
    for (int rt = 0; rt < 2; ++rt)
#pragma unroll
      for (int t = 0; t < 4; ++t) {
        int n = cb + t * 16 + lr;
        float fv = field[g * H_ + n];
        float bs = bt2[n];
#pragma unroll
        for (int v = 0; v < 8; ++v) {
          int m = rt * 16 + hi * 8 + v;
          bufZ[(size_t)m * H_ + n] = (_Float16)((acc[rt][t][v] + bs) * fv);
        }
      }
  }
  __syncthreads();

  // stage 2: h = relu(z @ wo1 + bo1) -> bufA (overwrite t1)
  {
    v8f acc[2][4];
#pragma unroll
    for (int rt = 0; rt < 2; ++rt)
#pragma unroll
      for (int t = 0; t < 4; ++t) acc[rt][t] = vzero8();
    gemm_k512_2x4(bufZ + (size_t)lr * H_, bufZ + (size_t)(16 + lr) * H_,
                  wo1T, cb, lr, hi, acc);
#pragma unroll
    for (int rt = 0; rt < 2; ++rt)
#pragma unroll
      for (int t = 0; t < 4; ++t) {
        int n = cb + t * 16 + lr;
        float bs = bo1[n];
#pragma unroll
        for (int v = 0; v < 8; ++v) {
          int m = rt * 16 + hi * 8 + v;
          bufA[(size_t)m * H_ + n] = (_Float16)fmaxf(acc[rt][t][v] + bs, 0.f);
        }
      }
  }
  __syncthreads();

  // stage 3: out = h @ wo2 + bo2   (zero-padded 512x16 B tile, waves 0..1)
  if (wave < 2) {
    v8f acc = vzero8();
    const _Float16* Arow = bufA + (size_t)(wave * 16 + lr) * H_;
    for (int k0 = 0; k0 < H_; k0 += 32) {
      v16h a = load_frag(Arow + k0, hi);
      v16h b = load_frag(wo2Tp + (size_t)lr * H_ + k0, hi);
      acc = wmma_f16(a, b, acc);
    }
    if (lr < OUT_) {
      float bs = bo2[lr];
#pragma unroll
      for (int v = 0; v < 8; ++v) {
        int row = r0 + wave * 16 + hi * 8 + v;
        out[(size_t)row * OUT_ + lr] = acc[v] + bs;
      }
    }
  }
}

// ---------------- host side ----------------
extern "C" void kernel_launch(void* const* d_in, const int* in_sizes, int n_in,
                              void* d_out, int out_size, void* d_ws, size_t ws_size,
                              hipStream_t stream) {
  (void)in_sizes; (void)n_in; (void)out_size; (void)ws_size;
  const float* known = (const float*)d_in[0];
  const float* nodes = (const float*)d_in[1];
  const int* esrc = (const int*)d_in[2];
  const int* edst = (const int*)d_in[3];
  const float* gw0 = (const float*)d_in[6];
  const float* gb0 = (const float*)d_in[7];
  const float* gw1 = (const float*)d_in[8];
  const float* gb1 = (const float*)d_in[9];
  const float* gw2 = (const float*)d_in[10];
  const float* gb2 = (const float*)d_in[11];
  const float* gwr = (const float*)d_in[12];
  const float* gbr = (const float*)d_in[13];
  const float* wt1 = (const float*)d_in[14];
  const float* bt1 = (const float*)d_in[15];
  const float* wt2 = (const float*)d_in[16];
  const float* bt2 = (const float*)d_in[17];
  const float* wo1 = (const float*)d_in[18];
  const float* bo1 = (const float*)d_in[19];
  const float* wo2 = (const float*)d_in[20];
  const float* bo2 = (const float*)d_in[21];
  float* out = (float*)d_out;

  char* ws = (char*)d_ws;
  size_t off = 0;
  auto alloc = [&](size_t bytes) -> void* {
    void* p = ws + off;
    off = (off + bytes + 255) & ~(size_t)255;
    return p;
  };
  float*     invdeg = (float*)alloc((size_t)NK_ * 4);
  float*     m0     = (float*)alloc((size_t)NK_ * 3 * 4);
  float*     mbuf   = (float*)alloc((size_t)NK_ * H_ * 4);     // 64 MB (reused)
  _Float16*  hA     = (_Float16*)alloc((size_t)NK_ * H_ * 2);  // h0 then h2
  _Float16*  hB     = (_Float16*)alloc((size_t)NK_ * H_ * 2);  // h1
  _Float16*  w1T    = (_Float16*)alloc((size_t)H_ * H_ * 2);
  _Float16*  w2T    = (_Float16*)alloc((size_t)H_ * H_ * 2);
  _Float16*  wt2T   = (_Float16*)alloc((size_t)H_ * H_ * 2);
  _Float16*  wo1T   = (_Float16*)alloc((size_t)H_ * H_ * 2);
  _Float16*  wo2Tp  = (_Float16*)alloc((size_t)16 * H_ * 2);
  float*     pooled = (float*)alloc((size_t)B_GR * H_ * 4);
  float*     field  = (float*)alloc((size_t)B_GR * H_ * 4);

  const int T = 256;
  // zero atomic accumulators
  zero_f32<<<(NK_ + T - 1) / T, T, 0, stream>>>(invdeg, NK_);
  zero_f32<<<(NK_ * 3 + T - 1) / T, T, 0, stream>>>(m0, (long)NK_ * 3);
  zero_f32<<<((long)NK_ * H_ + T - 1) / T, T, 0, stream>>>(mbuf, (long)NK_ * H_);

  // weight prep: transposed f16 copies (L2-resident operands for WMMA B)
  conv_transpose_f16<<<(H_ * H_) / T, T, 0, stream>>>(gw1, w1T);
  conv_transpose_f16<<<(H_ * H_) / T, T, 0, stream>>>(gw2, w2T);
  conv_transpose_f16<<<(H_ * H_) / T, T, 0, stream>>>(wt2, wt2T);
  conv_transpose_f16<<<(H_ * H_) / T, T, 0, stream>>>(wo1, wo1T);
  conv_wo2_pad<<<(16 * H_) / T, T, 0, stream>>>(wo2, wo2Tp);

  // degrees
  deg_count<<<(E_ + T - 1) / T, T, 0, stream>>>(edst, invdeg);
  deg_invert<<<(NK_ + T - 1) / T, T, 0, stream>>>(invdeg);

  // GCN layer 0 (F=3 dense)
  agg_edges_f3<<<(E_ + T - 1) / T, T, 0, stream>>>(known, esrc, edst, m0);
  layer0_dense<<<((long)NK_ * H_) / T, T, 0, stream>>>(m0, invdeg, gw0, gb0, hA);

  // GCN layer 1
  agg_edges_f16<<<E_, T, 0, stream>>>(hA, esrc, edst, mbuf);
  gcn_gemm_relu<<<NK_ / 32, T, 0, stream>>>(mbuf, invdeg, w1T, gb1, hB);

  // GCN layer 2 (reuse mbuf after re-zero, h2 -> hA)
  zero_f32<<<((long)NK_ * H_ + T - 1) / T, T, 0, stream>>>(mbuf, (long)NK_ * H_);
  agg_edges_f16<<<E_, T, 0, stream>>>(hB, esrc, edst, mbuf);
  gcn_gemm_relu<<<NK_ / 32, T, 0, stream>>>(mbuf, invdeg, w2T, gb2, hA);

  // per-graph readout -> field feats
  pool_mean<<<(B_GR * H_) / T, T, 0, stream>>>(hA, pooled);
  field_readout<<<(B_GR * H_) / T, T, 0, stream>>>(pooled, gwr, gbr, field);

  // fused trunk + broadcast multiply + out MLP (dominant cost, WMMA-bound)
  trunk_fused<<<NQ_ / 32, T, 2 * 32 * H_ * sizeof(_Float16), stream>>>(
      nodes, wt1, bt1, wt2T, bt2, field, wo1T, bo1, wo2Tp, bo2, out);
}
